// AttentionBlock_9242769622099
// MI455X (gfx1250) — matrix operand order repaired and checked by reading, not compiled, not run on hardware
//
#include <hip/hip_runtime.h>
#include <hip/hip_bf16.h>
#include <math.h>
#include <stdint.h>

// ---------------------------------------------------------------------------
// AttentionBlock for MI455X (gfx1250): GN -> QKV GEMM -> flash attention ->
// proj GEMM + residual.  Matmuls via v_wmma_f32_16x16x32_f16 (wave32);
// V tiles staged to LDS with GLOBAL_LOAD_ASYNC_TO_LDS_B128 (ASYNCcnt),
// double-buffered against the score WMMAs.  Weights pre-converted to f16
// once, so GEMM inner loops are pure b128-load -> WMMA.
// ---------------------------------------------------------------------------

typedef _Float16 h8  __attribute__((ext_vector_type(8)));
typedef _Float16 h16 __attribute__((ext_vector_type(16)));
typedef float    f8  __attribute__((ext_vector_type(8)));

#define CH   512
#define T3   1536          // 3*CH
#define TT   1024          // T = 32*32
#define NHD  8
#define HCH  64            // CH/NHD
#define VST  72            // V LDS row stride in halves (144B, 16B-aligned)

__device__ __forceinline__ f8 wmma16(h16 a, h16 b, f8 c) {
  return __builtin_amdgcn_wmma_f32_16x16x32_f16(false, a, false, b, (short)0, c,
                                                false, false);
}

// A-fragment (16x32, f16): lane m (m=lane&15, hh=lane>>4) holds row m,
// K = hh*16 .. hh*16+15 (contiguous halves).
__device__ __forceinline__ h16 loadA_f16(const _Float16* base, int stride) {
  int lane = threadIdx.x & 31, m = lane & 15, hh = lane >> 4;
  const _Float16* p = base + (size_t)m * stride + hh * 16;
  h8 lo = *(const h8*)(p);
  h8 hi = *(const h8*)(p + 8);
  h16 r;
#pragma unroll
  for (int j = 0; j < 8; ++j) { r[j] = lo[j]; r[j + 8] = hi[j]; }
  return r;
}

// B-fragment (32x16, f16) from a [N][K] (row = column-of-B) layout:
// lane n holds column n, K = hh*16 .. hh*16+15 (contiguous halves).
__device__ __forceinline__ h16 loadB_f16(const _Float16* base, int stride) {
  int lane = threadIdx.x & 31, n = lane & 15, hh = lane >> 4;
  const _Float16* p = base + (size_t)n * stride + hh * 16;
  h8 lo = *(const h8*)(p);
  h8 hi = *(const h8*)(p + 8);
  h16 r;
#pragma unroll
  for (int j = 0; j < 8; ++j) { r[j] = lo[j]; r[j + 8] = hi[j]; }
  return r;
}

// B-fragment for V^T gathered from a raw [s][c] LDS block (stride VST):
// lane n holds column c = (base offset)+n, K = s = hh*16 + j.
__device__ __forceinline__ h16 loadB_vT(const _Float16* vraw_c0) {
  int lane = threadIdx.x & 31, n = lane & 15, hh = lane >> 4;
  const _Float16* p = vraw_c0 + n + (size_t)(hh * 16) * VST;
  h16 r;
#pragma unroll
  for (int j = 0; j < 16; ++j) r[j] = p[(size_t)j * VST];
  return r;
}

// Async DMA one 32x64-half V block (rows s0..s0+31) into LDS [s][c].
// Each lane copies its own row as 8 x b128. Tracked by ASYNCcnt.
__device__ __forceinline__ void issue_v_async(const _Float16* vp, int s0,
                                              _Float16* dst) {
  const int lane = threadIdx.x & 31;
  const _Float16* g = vp + (size_t)(s0 + lane) * T3;
  unsigned d = (unsigned)(uintptr_t)(dst + (size_t)lane * VST);
#pragma unroll
  for (int ch = 0; ch < 8; ++ch) {
    asm volatile("global_load_async_to_lds_b128 %0, %1, off"
                 :: "v"(d + ch * 16),
                    "v"((unsigned long long)(uintptr_t)(g + ch * 8))
                 : "memory");
  }
}

// ---------------------------------------------------------------------------
// 0) One-shot fp32 -> f16 weight conversion (grid-stride).
// ---------------------------------------------------------------------------
__global__ void cvt_w_kernel(const float* __restrict__ in,
                             _Float16* __restrict__ out, int n) {
  for (int i = blockIdx.x * 256 + threadIdx.x; i < n; i += gridDim.x * 256)
    out[i] = (_Float16)in[i];
}

// ---------------------------------------------------------------------------
// 1) GroupNorm: x[b][c][t] fp32 -> h16t[b][t][c] f16.  One block per (b,g).
// ---------------------------------------------------------------------------
__global__ void gn_kernel(const float* __restrict__ x,
                          const float* __restrict__ w,
                          const float* __restrict__ bias,
                          _Float16* __restrict__ h16t) {
  const int b = blockIdx.x >> 5;   // 32 groups per batch
  const int g = blockIdx.x & 31;
  const int tid = threadIdx.x;
  const float* xp = x + ((size_t)b * CH + g * 16) * TT;   // 16 ch x 1024 t

  float s = 0.f, ss = 0.f;
  for (int e = tid; e < 16 * TT; e += 256) {
    float v = xp[e];
    s += v; ss += v * v;
  }
  __shared__ float red[512];
  red[tid] = s; red[256 + tid] = ss;
  __syncthreads();
  for (int off = 128; off > 0; off >>= 1) {
    if (tid < off) {
      red[tid]       += red[tid + off];
      red[256 + tid] += red[256 + tid + off];
    }
    __syncthreads();
  }
  const float inv_n = 1.0f / (16.0f * TT);
  const float mu  = red[0] * inv_n;
  const float var = red[256] * inv_n - mu * mu;
  const float rs  = rsqrtf(var + 1e-5f);

  for (int e = tid; e < 16 * TT; e += 256) {
    int lc = e >> 10;            // local channel 0..15
    int t  = e & (TT - 1);
    int c  = g * 16 + lc;
    float hn = (xp[e] - mu) * rs * w[c] + bias[c];
    h16t[((size_t)b * TT + t) * CH + c] = (_Float16)hn;
  }
}

// ---------------------------------------------------------------------------
// 2) QKV GEMM: qkvT[b][t][o] = sum_c qkv_w[o][c]*h[b][c][t] + qkv_b[o]
// grid (16 tBlk, 24 oBlk, 8 b), block 128 (4 waves). Wave: 16 o x 64 t tile.
// Software-pipelined one K-step ahead; all-f16 fragment loads.
// ---------------------------------------------------------------------------
__global__ void qkv_gemm(const _Float16* __restrict__ w16,
                         const float* __restrict__ qkv_b,
                         const _Float16* __restrict__ h16t,
                         _Float16* __restrict__ qkvT) {
  const int b    = blockIdx.z;
  const int wave = threadIdx.x >> 5;
  const int o0   = (blockIdx.y * 4 + wave) * 16;
  const int t0   = blockIdx.x * 64;
  const _Float16* A  = w16 + (size_t)o0 * CH;
  const _Float16* Bb = h16t + ((size_t)b * TT + t0) * CH;

  f8 acc[4] = {};
  h16 af  = loadA_f16(A, CH);
  h16 bf0 = loadB_f16(Bb,                   CH);
  h16 bf1 = loadB_f16(Bb + (size_t)16 * CH, CH);
  h16 bf2 = loadB_f16(Bb + (size_t)32 * CH, CH);
  h16 bf3 = loadB_f16(Bb + (size_t)48 * CH, CH);
  for (int k = 0; k < CH; k += 32) {
    const int kn = (k + 32) & (CH - 1);
    h16 an  = loadA_f16(A + kn, CH);
    h16 b0n = loadB_f16(Bb + kn,                   CH);
    h16 b1n = loadB_f16(Bb + (size_t)16 * CH + kn, CH);
    h16 b2n = loadB_f16(Bb + (size_t)32 * CH + kn, CH);
    h16 b3n = loadB_f16(Bb + (size_t)48 * CH + kn, CH);
    acc[0] = wmma16(af, bf0, acc[0]);
    acc[1] = wmma16(af, bf1, acc[1]);
    acc[2] = wmma16(af, bf2, acc[2]);
    acc[3] = wmma16(af, bf3, acc[3]);
    af = an; bf0 = b0n; bf1 = b1n; bf2 = b2n; bf3 = b3n;
  }
  const int lane = threadIdx.x & 31, n = lane & 15, hh = lane >> 4;
#pragma unroll
  for (int r = 0; r < 8; ++r) {
    const int o = o0 + r + hh * 8;
    const float bo = qkv_b[o];
#pragma unroll
    for (int q = 0; q < 4; ++q) {
      const int t = t0 + q * 16 + n;
      qkvT[((size_t)b * TT + t) * T3 + o] = (_Float16)(acc[q][r] + bo);
    }
  }
}

// ---------------------------------------------------------------------------
// 3) Flash attention: per (bh, 16-row t-tile) wave. S = (q sc)^T (k sc),
// online softmax, acc += P V^T.  V blocks double-buffered via async DMA.
// grid (64 bh, 16 tGrp), block 128 (4 waves). Writes aT[b][t][c] f16.
// ---------------------------------------------------------------------------
__global__ void attn_kernel(const _Float16* __restrict__ qkvT,
                            _Float16* __restrict__ aT) {
  const int bh = blockIdx.x;
  const int b  = bh >> 3;
  const int hd = bh & 7;
  const int wave = threadIdx.x >> 5;
  const int lane = threadIdx.x & 31;
  const int t0 = (blockIdx.y * 4 + wave) * 16;

  __shared__ __align__(16) _Float16 Pl[4][16][40];
  __shared__ __align__(16) _Float16 Vraw[4][2][32][VST];   // [s][c] raw blocks

  const _Float16* base = qkvT + (size_t)b * TT * T3;
  const _Float16* qp = base + hd * 192;
  const _Float16* kp = base + hd * 192 + 64;
  const _Float16* vp = base + hd * 192 + 128;

  // Q fragments for this t-tile (K = head-channel, 0..63 -> two frags)
  const h16 aq0 = loadA_f16(qp + (size_t)t0 * T3,      T3);
  const h16 aq1 = loadA_f16(qp + (size_t)t0 * T3 + 32, T3);

  f8 acc[4] = {};
  float m[8], l[8];
#pragma unroll
  for (int r = 0; r < 8; ++r) { m[r] = -3.0e38f; l[r] = 0.f; }

  const float sc = 0.125f;           // (1/sqrt(sqrt(64)))^2
  const int n = lane & 15, hh = lane >> 4;

  // prologue: async-stage V block 0 into buffer 0
  issue_v_async(vp, 0, &Vraw[wave][0][0][0]);

  int ib = 0;
  for (int s0 = 0; s0 < TT; s0 += 32, ib ^= 1) {
    const bool more = (s0 + 32) < TT;
    // WAR guard: prior ds reads of the buffer we are about to overwrite
    asm volatile("s_wait_dscnt 0x0" ::: "memory");
    if (more) {
      issue_v_async(vp, s0 + 32, &Vraw[wave][ib ^ 1][0][0]);
      __builtin_prefetch(kp + (size_t)(s0 + 32 + lane) * T3, 0, 1);
    }

    // ---- scores: two 16x16 D tiles covering s0..s0+31 ----
    f8 sA = {}, sB = {};
    sA = wmma16(aq0, loadB_f16(kp + (size_t)s0        * T3,      T3), sA);
    sA = wmma16(aq1, loadB_f16(kp + (size_t)s0        * T3 + 32, T3), sA);
    sB = wmma16(aq0, loadB_f16(kp + (size_t)(s0 + 16) * T3,      T3), sB);
    sB = wmma16(aq1, loadB_f16(kp + (size_t)(s0 + 16) * T3 + 32, T3), sB);

    // ---- online softmax (rows r+8*hh live in register r) ----
#pragma unroll
    for (int r = 0; r < 8; ++r) {
      float e0 = sA[r] * sc, e1 = sB[r] * sc;
      float mx = fmaxf(e0, e1);
      for (int off = 8; off > 0; off >>= 1)
        mx = fmaxf(mx, __shfl_xor(mx, off, 32));
      const float mnew = fmaxf(m[r], mx);
      const float corr = __expf(m[r] - mnew);
      const float p0 = __expf(e0 - mnew);
      const float p1 = __expf(e1 - mnew);
      float rsum = p0 + p1;
      for (int off = 8; off > 0; off >>= 1)
        rsum += __shfl_xor(rsum, off, 32);
      l[r] = l[r] * corr + rsum;
      m[r] = mnew;
#pragma unroll
      for (int cc = 0; cc < 4; ++cc) acc[cc][r] *= corr;
      const int row = r + hh * 8;
      Pl[wave][row][n]      = (_Float16)p0;
      Pl[wave][row][n + 16] = (_Float16)p1;
    }

    // current V buffer complete: 8 newer asyncs may stay in flight
    if (more) asm volatile("s_wait_asynccnt 0x8" ::: "memory");
    else      asm volatile("s_wait_asynccnt 0x0" ::: "memory");
    __syncthreads();

    // ---- acc += P (16x32) x V^T (32x16 per c-subtile) ----
    const h16 aP = loadA_f16(&Pl[wave][0][0], 40);
#pragma unroll
    for (int cc = 0; cc < 4; ++cc)
      acc[cc] = wmma16(aP, loadB_vT(&Vraw[wave][ib][0][cc * 16]), acc[cc]);
    __syncthreads();
  }

  // ---- epilogue: aT[b][t][hd*64 + c] = acc / l ----
#pragma unroll
  for (int r = 0; r < 8; ++r) {
    const int t = t0 + r + hh * 8;
    const float inv_l = 1.0f / l[r];
#pragma unroll
    for (int cc = 0; cc < 4; ++cc) {
      const int c = hd * HCH + cc * 16 + n;
      aT[((size_t)b * TT + t) * CH + c] = (_Float16)(acc[cc][r] * inv_l);
    }
  }
}

// ---------------------------------------------------------------------------
// 4) Proj GEMM + residual: out[b][o][t] = x[b][o][t] + proj_w @ a + proj_b
// grid (16 tBlk, 8 oBlk, 8 b), block 128 (4 waves). Wave: 16 o x 64 t tile.
// ---------------------------------------------------------------------------
__global__ void proj_gemm(const _Float16* __restrict__ w16,
                          const float* __restrict__ proj_b,
                          const _Float16* __restrict__ aT,
                          const float* __restrict__ x,
                          float* __restrict__ out) {
  const int b    = blockIdx.z;
  const int wave = threadIdx.x >> 5;
  const int o0   = (blockIdx.y * 4 + wave) * 16;
  const int t0   = blockIdx.x * 64;
  const _Float16* A  = w16 + (size_t)o0 * CH;
  const _Float16* Bb = aT + ((size_t)b * TT + t0) * CH;

  f8 acc[4] = {};
  h16 af  = loadA_f16(A, CH);
  h16 bf0 = loadB_f16(Bb,                   CH);
  h16 bf1 = loadB_f16(Bb + (size_t)16 * CH, CH);
  h16 bf2 = loadB_f16(Bb + (size_t)32 * CH, CH);
  h16 bf3 = loadB_f16(Bb + (size_t)48 * CH, CH);
  for (int k = 0; k < CH; k += 32) {
    const int kn = (k + 32) & (CH - 1);
    h16 an  = loadA_f16(A + kn, CH);
    h16 b0n = loadB_f16(Bb + kn,                   CH);
    h16 b1n = loadB_f16(Bb + (size_t)16 * CH + kn, CH);
    h16 b2n = loadB_f16(Bb + (size_t)32 * CH + kn, CH);
    h16 b3n = loadB_f16(Bb + (size_t)48 * CH + kn, CH);
    acc[0] = wmma16(af, bf0, acc[0]);
    acc[1] = wmma16(af, bf1, acc[1]);
    acc[2] = wmma16(af, bf2, acc[2]);
    acc[3] = wmma16(af, bf3, acc[3]);
    af = an; bf0 = b0n; bf1 = b1n; bf2 = b2n; bf3 = b3n;
  }
  const int lane = threadIdx.x & 31, n = lane & 15, hh = lane >> 4;
#pragma unroll
  for (int r = 0; r < 8; ++r) {
    const int o = o0 + r + hh * 8;
    const float bo = proj_b[o];
#pragma unroll
    for (int q = 0; q < 4; ++q) {
      const int t = t0 + q * 16 + n;
      const size_t idx = ((size_t)b * CH + o) * TT + t;
      out[idx] = x[idx] + acc[q][r] + bo;
    }
  }
}

// ---------------------------------------------------------------------------
extern "C" void kernel_launch(void* const* d_in, const int* in_sizes, int n_in,
                              void* d_out, int out_size, void* d_ws, size_t ws_size,
                              hipStream_t stream) {
  const float* x      = (const float*)d_in[0];
  const float* gn_w   = (const float*)d_in[1];
  const float* gn_b   = (const float*)d_in[2];
  const float* qkv_w  = (const float*)d_in[3];
  const float* qkv_b  = (const float*)d_in[4];
  const float* proj_w = (const float*)d_in[5];
  const float* proj_b = (const float*)d_in[6];
  float* out = (float*)d_out;

  // workspace layout (f16): h16t 8MB | qkvT 24MB | aT 8MB | w16 1.5MB | 0.5MB
  _Float16* h16t  = (_Float16*)d_ws;
  _Float16* qkvT  = h16t + (size_t)8 * TT * CH;
  _Float16* aT    = qkvT + (size_t)8 * TT * T3;
  _Float16* qw16  = aT   + (size_t)8 * TT * CH;
  _Float16* pw16  = qw16 + (size_t)T3 * CH;

  cvt_w_kernel<<<768, 256, 0, stream>>>(qkv_w, qw16, T3 * CH);
  cvt_w_kernel<<<256, 256, 0, stream>>>(proj_w, pw16, CH * CH);
  gn_kernel<<<256, 256, 0, stream>>>(x, gn_w, gn_b, h16t);
  qkv_gemm<<<dim3(16, 24, 8), 128, 0, stream>>>(qw16, qkv_b, h16t, qkvT);
  attn_kernel<<<dim3(64, 16), 128, 0, stream>>>(qkvT, aT);
  proj_gemm<<<dim3(16, 8, 8), 128, 0, stream>>>(pw16, proj_b, aT, x, out);
}
